// INTFeedForward_87789131530597
// MI455X (gfx1250) — 8-bit matrix operand order repaired and checked by reading, compiled, not run on hardware
//
#include <hip/hip_runtime.h>
#include <hip/hip_bf16.h>

#define TOKENS   8192
#define DIM      1280
#define TWOIN    10240
#define INNER    5120
#define A1f      3.0e-5f
#define B1f      0.05f
#define INV_SHf  50.0f
#define A2f      7.5e-4f
#define B2f      0.05f

typedef __attribute__((ext_vector_type(8))) int v8i;

#define WMMA_IU8(a, b, c) \
    __builtin_amdgcn_wmma_i32_16x16x64_iu8(true, (a), true, (b), (c), false, false)

__global__ void cvt_f32_to_i8(const float4* __restrict__ in,
                              unsigned* __restrict__ out, int n4) {
    int i = blockIdx.x * blockDim.x + threadIdx.x;
    if (i >= n4) return;
    float4 v = in[i];
    int b0 = __float2int_rn(v.x);
    int b1 = __float2int_rn(v.y);
    int b2 = __float2int_rn(v.z);
    int b3 = __float2int_rn(v.w);
    out[i] = ( (unsigned)(b0 & 0xFF)       ) |
             ( (unsigned)(b1 & 0xFF) <<  8 ) |
             ( (unsigned)(b2 & 0xFF) << 16 ) |
             ( (unsigned)(b3 & 0xFF) << 24 );
}

__device__ __forceinline__ v8i load_a_tile(const char* __restrict__ rowptr) {
    int4 c0 = *(const int4*)(rowptr +  0);
    int4 c1 = *(const int4*)(rowptr + 32);
    v8i a;
    a[0] = c0.x; a[1] = c0.y; a[2] = c0.z; a[3] = c0.w;
    a[4] = c1.x; a[5] = c1.y; a[6] = c1.z; a[7] = c1.w;
    return a;
}
__device__ __forceinline__ v8i load_b_tile(const char* __restrict__ rowptr) {
    int4 c0 = *(const int4*)(rowptr +  0);
    int4 c1 = *(const int4*)(rowptr + 32);
    v8i b;
    b[0] = c0.x; b[1] = c0.y; b[2] = c0.z; b[3] = c0.w;
    b[4] = c1.x; b[5] = c1.y; b[6] = c1.z; b[7] = c1.w;
    return b;
}

__global__ void __launch_bounds__(256)
geglu_gemm1(const char* __restrict__ x8, const char* __restrict__ w18,
            const float* __restrict__ b1, char* __restrict__ q8) {
    const int lane = threadIdx.x & 31;
    const int wave = threadIdx.x >> 5;
    const int MBLK = TOKENS / 64;
    const int blockRow = blockIdx.x % MBLK;
    const int blockCol = blockIdx.x / MBLK;
    const int mt = blockRow * 2 + (wave >> 2);
    const int nt = blockCol * 4 + (wave & 3);

    const int laneHalf = lane >> 4;
    const int row0 = mt * 32 + (lane & 15);
    const int col  = nt * 16 + (lane & 15);

    const char* aptr0 = x8 + (size_t)row0 * DIM + (laneHalf << 4);
    const char* aptr1 = aptr0 + (size_t)16 * DIM;
    const char* bhptr = w18 + (size_t)col * DIM + (laneHalf << 4);
    const char* bgptr = w18 + (size_t)(col + INNER) * DIM + (laneHalf << 4);

    v8i ah0 = {}, ag0 = {}, ah1 = {}, ag1 = {};
#pragma unroll 2
    for (int k0 = 0; k0 < DIM; k0 += 64) {
        v8i a0 = load_a_tile(aptr0 + k0);
        v8i a1 = load_a_tile(aptr1 + k0);
        v8i bh = load_b_tile(bhptr + k0);
        v8i bg = load_b_tile(bgptr + k0);
        ah0 = WMMA_IU8(a0, bh, ah0);
        ag0 = WMMA_IU8(a0, bg, ag0);
        ah1 = WMMA_IU8(a1, bh, ah1);
        ag1 = WMMA_IU8(a1, bg, ag1);
    }

    const float bias_h = b1[col] * B1f;
    const float bias_g = b1[col + INNER] * B1f;
    const int tokBase0 = mt * 32 + (laneHalf << 3);
    const int tokBase1 = tokBase0 + 16;
#pragma unroll
    for (int r = 0; r < 8; ++r) {
        float h0 = (float)ah0[r] * A1f + bias_h;
        float g0 = (float)ag0[r] * A1f + bias_g;
        float y0 = h0 * (0.5f * g0 * (1.0f + erff(g0 * 0.70710678118654752440f)));
        float q0 = fminf(fmaxf(rintf(y0 * INV_SHf), -128.0f), 127.0f);
        q8[(size_t)(tokBase0 + r) * INNER + col] = (char)(int)q0;

        float h1 = (float)ah1[r] * A1f + bias_h;
        float g1 = (float)ag1[r] * A1f + bias_g;
        float y1 = h1 * (0.5f * g1 * (1.0f + erff(g1 * 0.70710678118654752440f)));
        float q1 = fminf(fmaxf(rintf(y1 * INV_SHf), -128.0f), 127.0f);
        q8[(size_t)(tokBase1 + r) * INNER + col] = (char)(int)q1;
    }
}

__global__ void __launch_bounds__(256)
out_gemm2(const char* __restrict__ q8, const char* __restrict__ w28,
          const float* __restrict__ b2, float* __restrict__ out) {
    const int lane = threadIdx.x & 31;
    const int wave = threadIdx.x >> 5;
    const int MBLK = TOKENS / 64;
    const int blockRow = blockIdx.x % MBLK;
    const int blockCol = blockIdx.x / MBLK;
    const int mt = blockRow * 2 + (wave >> 2);
    const int nt = blockCol * 4 + (wave & 3);

    const int laneHalf = lane >> 4;
    const int row0 = mt * 32 + (lane & 15);
    const int col0 = nt * 32 + (lane & 15);
    const int col1 = col0 + 16;

    const char* aptr0 = q8 + (size_t)row0 * INNER + (laneHalf << 4);
    const char* aptr1 = aptr0 + (size_t)16 * INNER;
    const char* bptr0 = w28 + (size_t)col0 * INNER + (laneHalf << 4);
    const char* bptr1 = w28 + (size_t)col1 * INNER + (laneHalf << 4);

    v8i acc00 = {}, acc01 = {}, acc10 = {}, acc11 = {};
#pragma unroll 2
    for (int k0 = 0; k0 < INNER; k0 += 64) {
        v8i a0 = load_a_tile(aptr0 + k0);
        v8i a1 = load_a_tile(aptr1 + k0);
        v8i b0 = load_b_tile(bptr0 + k0);
        v8i b1v = load_b_tile(bptr1 + k0);
        acc00 = WMMA_IU8(a0, b0,  acc00);
        acc01 = WMMA_IU8(a0, b1v, acc01);
        acc10 = WMMA_IU8(a1, b0,  acc10);
        acc11 = WMMA_IU8(a1, b1v, acc11);
    }

    const float bias0 = b2[col0] * B2f;
    const float bias1 = b2[col1] * B2f;
    const int tokBase0 = mt * 32 + (laneHalf << 3);
    const int tokBase1 = tokBase0 + 16;
#pragma unroll
    for (int r = 0; r < 8; ++r) {
        float v00 = fminf(fmaxf(rintf((float)acc00[r] * A2f + bias0), -128.0f), 127.0f);
        float v01 = fminf(fmaxf(rintf((float)acc01[r] * A2f + bias1), -128.0f), 127.0f);
        float v10 = fminf(fmaxf(rintf((float)acc10[r] * A2f + bias0), -128.0f), 127.0f);
        float v11 = fminf(fmaxf(rintf((float)acc11[r] * A2f + bias1), -128.0f), 127.0f);
        out[(size_t)(tokBase0 + r) * DIM + col0] = v00;
        out[(size_t)(tokBase0 + r) * DIM + col1] = v01;
        out[(size_t)(tokBase1 + r) * DIM + col0] = v10;
        out[(size_t)(tokBase1 + r) * DIM + col1] = v11;
    }
}

extern "C" void kernel_launch(void* const* d_in, const int* in_sizes, int n_in,
                              void* d_out, int out_size, void* d_ws, size_t ws_size,
                              hipStream_t stream) {
    const float* x  = (const float*)d_in[0];
    const float* W1 = (const float*)d_in[1];
    const float* b1 = (const float*)d_in[2];
    const float* W2 = (const float*)d_in[3];
    const float* b2 = (const float*)d_in[4];
    float* out = (float*)d_out;

    char* ws  = (char*)d_ws;
    char* x8  = ws;
    char* w18 = ws + (size_t)10485760;
    char* w28 = ws + (size_t)23592960;
    char* q8  = ws + (size_t)30146560;

    const int nx4  = (TOKENS * DIM) / 4;
    const int nw14 = (TWOIN * DIM) / 4;
    const int nw24 = (DIM * INNER) / 4;

    cvt_f32_to_i8<<<(nx4  + 255) / 256, 256, 0, stream>>>(
        (const float4*)x,  (unsigned*)x8,  nx4);
    cvt_f32_to_i8<<<(nw14 + 255) / 256, 256, 0, stream>>>(
        (const float4*)W1, (unsigned*)w18, nw14);
    cvt_f32_to_i8<<<(nw24 + 255) / 256, 256, 0, stream>>>(
        (const float4*)W2, (unsigned*)w28, nw24);

    geglu_gemm1<<<10240, 256, 0, stream>>>(x8, w18, b1, q8);

    out_gemm2<<<1280, 256, 0, stream>>>(q8, w28, b2, out);
}
